// MyModel_56444460204607
// MI455X (gfx1250) — matrix operand order repaired and checked by reading, not compiled, not run on hardware
//
#include <hip/hip_runtime.h>
#include <hip/hip_bf16.h>

// ---------------------------------------------------------------------------
// Types for CDNA5 WMMA
// ---------------------------------------------------------------------------
typedef __attribute__((ext_vector_type(16))) __bf16    v16bf;
typedef __attribute__((ext_vector_type(8)))  float     v8f;
typedef __attribute__((ext_vector_type(4)))  unsigned  uint4v;
typedef __attribute__((ext_vector_type(2)))  unsigned  uint2v;
typedef __attribute__((ext_vector_type(4)))  float     float4v;
typedef __attribute__((ext_vector_type(4)))  int       int4v;

union FragBF {
    v16bf  v;      // 16 bf16 = 8 VGPRs (WMMA A/B operand, wave32)
    uint4v u[2];   // two 16-byte chunks for LDS b128 loads
};

// ---------------------------------------------------------------------------
// Async global->LDS copy (CDNA5 GLOBAL_LOAD_ASYNC_TO_LDS_B128, ASYNCcnt).
// Builtin signature (from hipcc diagnostic): (v4i AS1*, v4i AS3*, Ii, Ii).
// Falls back to a synchronous load + ds_store if the builtin is unavailable.
// ---------------------------------------------------------------------------
#if __has_builtin(__builtin_amdgcn_global_load_async_to_lds_b128)
#define USE_ASYNC_LDS 1
#else
#define USE_ASYNC_LDS 0
#endif

typedef __attribute__((address_space(1))) int4v gas_int4v;  // global
typedef __attribute__((address_space(3))) int4v las_int4v;  // LDS

static __device__ __forceinline__ void async_cp16(const void* gsrc, void* ldst) {
#if USE_ASYNC_LDS
    __builtin_amdgcn_global_load_async_to_lds_b128(
        (gas_int4v*)gsrc, (las_int4v*)ldst, 0, 0);
#else
    *(uint4v*)ldst = *(const uint4v*)gsrc;
#endif
}

static __device__ __forceinline__ void wait_async0() {
#if USE_ASYNC_LDS
#if __has_builtin(__builtin_amdgcn_s_wait_asynccnt)
    __builtin_amdgcn_s_wait_asynccnt(0);
#else
    asm volatile("s_wait_asynccnt 0x0" ::: "memory");
#endif
#endif
}

// ---------------------------------------------------------------------------
// fp32 -> bf16 round-to-nearest-even (bit manipulation; no libdevice dep)
// ---------------------------------------------------------------------------
static __device__ __forceinline__ unsigned short bf16_rne(float f) {
    unsigned u = __float_as_uint(f);
    u += 0x7FFFu + ((u >> 16) & 1u);
    return (unsigned short)(u >> 16);
}
static __device__ __forceinline__ float bf16_to_f(unsigned short h) {
    return __uint_as_float(((unsigned)h) << 16);
}

// Split 16 fp32 into hi/lo bf16 pairs, packed 2-per-dword into LDS.
static __device__ __forceinline__ void split_store16(const float* vals,
                                                     __bf16* hi, __bf16* lo) {
    unsigned* dh = (unsigned*)hi;
    unsigned* dl = (unsigned*)lo;
#pragma unroll
    for (int p = 0; p < 8; ++p) {
        float f0 = vals[2 * p], f1 = vals[2 * p + 1];
        unsigned short h0 = bf16_rne(f0), h1 = bf16_rne(f1);
        float r0 = f0 - bf16_to_f(h0);
        float r1 = f1 - bf16_to_f(h1);
        unsigned short l0 = bf16_rne(r0), l1 = bf16_rne(r1);
        dh[p] = (unsigned)h0 | ((unsigned)h1 << 16);
        dl[p] = (unsigned)l0 | ((unsigned)l1 << 16);
    }
}

static __device__ __forceinline__ void fetch16(const float* __restrict__ src,
                                               float* vals) {
#pragma unroll
    for (int i = 0; i < 16; i += 4) {
        float4v f = *reinterpret_cast<const float4v*>(src + i);
        vals[i]     = f[0];
        vals[i + 1] = f[1];
        vals[i + 2] = f[2];
        vals[i + 3] = f[3];
    }
}

// ---------------------------------------------------------------------------
// Fragment loads from LDS tiles (tile = [128 rows][32 K] bf16, row-major).
// Layouts per CDNA5 ISA 7.12.2:
//   A 16x32:  lanes 0-15 hold K=0..7 (V0-3) and K=16..23 (V4-7);
//             lanes 16-31 hold K=8..15 and K=24..31.
//   B 32x16:  lanes 0-15 hold K=0..7 and K=16..23; lanes 16-31 hold K=8..15 and K=24..31.
// ---------------------------------------------------------------------------
static __device__ __forceinline__ FragBF load_fragA(const __bf16 (*T)[32],
                                                    int row, int lane) {
    FragBF f;
    const int kg = (lane >> 4) << 3;  // 0 or 8
    f.u[0] = *reinterpret_cast<const uint4v*>(&T[row][kg]);
    f.u[1] = *reinterpret_cast<const uint4v*>(&T[row][kg + 16]);
    return f;
}
static __device__ __forceinline__ FragBF load_fragB(const __bf16 (*T)[32],
                                                    int row, int lane) {
    FragBF f;
    const int kg = (lane >> 4) << 3;  // 0 or 8
    f.u[0] = *reinterpret_cast<const uint4v*>(&T[row][kg]);
    f.u[1] = *reinterpret_cast<const uint4v*>(&T[row][kg + 16]);
    return f;
}

// ===========================================================================
// FAST PATH: operands pre-split to hi/lo bf16 in workspace. GEMM staging is a
// pure 16B/lane async DMA into double-buffered LDS; inner loop = WMMA only.
// Workgroup tile 128x128, 8 waves (2M x 4N), wave tile 64x32, K step 32,
// 24 x v_wmma_f32_16x16x32_bf16 per step (hi*hi + lo*hi + hi*lo).
// ===========================================================================
__global__ __launch_bounds__(256) void gemm_nt_presplit(
    const __bf16* __restrict__ Ahi, const __bf16* __restrict__ Alo,
    const __bf16* __restrict__ Bhi, const __bf16* __restrict__ Blo,
    const float* __restrict__ bias, float* __restrict__ C,
    int M, int Nout, int K)
{
    __shared__ __align__(16) __bf16 As_hi[2][128][32];
    __shared__ __align__(16) __bf16 As_lo[2][128][32];
    __shared__ __align__(16) __bf16 Bs_hi[2][128][32];
    __shared__ __align__(16) __bf16 Bs_lo[2][128][32];

    const int t     = threadIdx.x;
    const int lane  = t & 31;
    const int w     = t >> 5;
    const int m0    = blockIdx.x * 128;   // x fastest: blocks sharing a B strip run together
    const int n0    = blockIdx.y * 128;
    const int waveM = (w >> 2) * 64;
    const int waveN = (w & 3) * 32;
    const int rowL  = lane & 15;

    // Staging: each thread DMAs one 16B chunk in rows r0 and r0+64 of each tile.
    const int o0 = t << 4;            // byte offset in 8KB tile
    const int r0 = o0 >> 6;           // 0..63
    const int c0 = (o0 & 63) >> 1;    // bf16 column: 0,8,16,24

    v8f acc[4][2];
#pragma unroll
    for (int j = 0; j < 2; ++j) {
        const int col = n0 + waveN + j * 16 + rowL;
        const float bz = (col < Nout) ? bias[col] : 0.0f;
        v8f init = {bz, bz, bz, bz, bz, bz, bz, bz};
#pragma unroll
        for (int i = 0; i < 4; ++i) acc[i][j] = init;
    }

    const size_t aOff0 = (size_t)(m0 + r0) * K + c0;
    const size_t aOff1 = (size_t)(m0 + r0 + 64) * K + c0;
    const bool   bOK0  = (n0 + r0) < Nout;
    const bool   bOK1  = (n0 + r0 + 64) < Nout;
    const size_t bOff0 = (size_t)(bOK0 ? (n0 + r0) : 0) * K + c0;
    const size_t bOff1 = (size_t)(bOK1 ? (n0 + r0 + 64) : 0) * K + c0;

#define STAGE_TILES(buf, k0)                                                     \
    do {                                                                         \
        async_cp16(Ahi + aOff0 + (k0), &As_hi[buf][r0][c0]);                     \
        async_cp16(Ahi + aOff1 + (k0), &As_hi[buf][r0 + 64][c0]);                \
        async_cp16(Alo + aOff0 + (k0), &As_lo[buf][r0][c0]);                     \
        async_cp16(Alo + aOff1 + (k0), &As_lo[buf][r0 + 64][c0]);                \
        if (bOK0) {                                                              \
            async_cp16(Bhi + bOff0 + (k0), &Bs_hi[buf][r0][c0]);                 \
            async_cp16(Blo + bOff0 + (k0), &Bs_lo[buf][r0][c0]);                 \
        }                                                                        \
        if (bOK1) {                                                              \
            async_cp16(Bhi + bOff1 + (k0), &Bs_hi[buf][r0 + 64][c0]);            \
            async_cp16(Blo + bOff1 + (k0), &Bs_lo[buf][r0 + 64][c0]);            \
        }                                                                        \
    } while (0)

    // ---- prologue: tile 0 -> buffer 0 ----
    STAGE_TILES(0, 0);
    wait_async0();
    __syncthreads();

    const int nk = K >> 5;
    for (int kt = 0; kt < nk; ++kt) {
        const int  cur  = kt & 1;
        const bool more = (kt + 1) < nk;

        // 1) Launch DMA of the next K-tile into the other buffer (overlaps WMMA).
        if (more) STAGE_TILES(cur ^ 1, (kt + 1) << 5);

        // 2) Compute on the current buffer.
        FragBF ah[4], al[4], bh[2], bl[2];
#pragma unroll
        for (int i = 0; i < 4; ++i) {
            const int r = waveM + i * 16 + rowL;
            ah[i] = load_fragA(As_hi[cur], r, lane);
            al[i] = load_fragA(As_lo[cur], r, lane);
        }
#pragma unroll
        for (int j = 0; j < 2; ++j) {
            const int r = waveN + j * 16 + rowL;
            bh[j] = load_fragB(Bs_hi[cur], r, lane);
            bl[j] = load_fragB(Bs_lo[cur], r, lane);
        }
#pragma unroll
        for (int i = 0; i < 4; ++i)
#pragma unroll
            for (int j = 0; j < 2; ++j) {
                acc[i][j] = __builtin_amdgcn_wmma_f32_16x16x32_bf16(
                    false, ah[i].v, false, bh[j].v, (short)0, acc[i][j], false, false);
                acc[i][j] = __builtin_amdgcn_wmma_f32_16x16x32_bf16(
                    false, al[i].v, false, bh[j].v, (short)0, acc[i][j], false, false);
                acc[i][j] = __builtin_amdgcn_wmma_f32_16x16x32_bf16(
                    false, ah[i].v, false, bl[j].v, (short)0, acc[i][j], false, false);
            }

        // 3) DMA for the next buffer must land before anyone reads it.
        wait_async0();
        __syncthreads();
    }
#undef STAGE_TILES

    // ---- epilogue ----
    const int rsel = (lane >> 4) << 3;  // 0 or 8
#pragma unroll
    for (int i = 0; i < 4; ++i)
#pragma unroll
        for (int j = 0; j < 2; ++j) {
            const int col = n0 + waveN + j * 16 + rowL;
            if (col < Nout) {
#pragma unroll
                for (int v = 0; v < 8; ++v) {
                    const int row = m0 + waveM + i * 16 + rsel + v;
                    C[(size_t)row * Nout + col] = acc[i][j][v];
                }
            }
        }
}

// ---------------------------------------------------------------------------
// Pre-pass: split fp32 array into hi/lo bf16 arrays (bandwidth bound).
// ---------------------------------------------------------------------------
__global__ __launch_bounds__(256) void split_pair_kernel(
    const float* __restrict__ src, __bf16* __restrict__ hi,
    __bf16* __restrict__ lo, long n)
{
    const long i = ((long)blockIdx.x * 256 + threadIdx.x) * 4;
    if (i + 3 < n) {
        float4v f = *reinterpret_cast<const float4v*>(src + i);
        unsigned hp[2], lp[2];
#pragma unroll
        for (int p = 0; p < 2; ++p) {
            unsigned short h0 = bf16_rne(f[2 * p]),  h1 = bf16_rne(f[2 * p + 1]);
            unsigned short l0 = bf16_rne(f[2 * p] - bf16_to_f(h0));
            unsigned short l1 = bf16_rne(f[2 * p + 1] - bf16_to_f(h1));
            hp[p] = (unsigned)h0 | ((unsigned)h1 << 16);
            lp[p] = (unsigned)l0 | ((unsigned)l1 << 16);
        }
        *reinterpret_cast<uint2v*>(hi + i) = uint2v{hp[0], hp[1]};
        *reinterpret_cast<uint2v*>(lo + i) = uint2v{lp[0], lp[1]};
    } else {
        for (long k = i; k < n; ++k) {
            float fv = src[k];
            unsigned short h = bf16_rne(fv);
            *(unsigned short*)(hi + k) = h;
            *(unsigned short*)(lo + k) = bf16_rne(fv - bf16_to_f(h));
        }
    }
}

// ---------------------------------------------------------------------------
// LayerNorm over D=2048, one block (256 threads) per row, hi/lo bf16 output.
// ---------------------------------------------------------------------------
__global__ __launch_bounds__(256) void layernorm_2048_split(
    const float* __restrict__ h, const float* __restrict__ gamma,
    const float* __restrict__ beta, __bf16* __restrict__ hn_hi,
    __bf16* __restrict__ hn_lo)
{
    const int row = blockIdx.x;
    const int t   = threadIdx.x;
    const float* hr = h + (size_t)row * 2048;
    const int base = t * 8;

    float v[8];
    {
        float4v f0 = *reinterpret_cast<const float4v*>(hr + base);
        float4v f1 = *reinterpret_cast<const float4v*>(hr + base + 4);
#pragma unroll
        for (int i = 0; i < 4; ++i) { v[i] = f0[i]; v[4 + i] = f1[i]; }
    }
    float s = 0.0f, ss = 0.0f;
#pragma unroll
    for (int i = 0; i < 8; ++i) { s += v[i]; ss += v[i] * v[i]; }

    __shared__ float r1[256], r2[256];
    r1[t] = s; r2[t] = ss;
    __syncthreads();
    for (int off = 128; off > 0; off >>= 1) {
        if (t < off) { r1[t] += r1[t + off]; r2[t] += r2[t + off]; }
        __syncthreads();
    }
    const float mu  = r1[0] * (1.0f / 2048.0f);
    const float var = r2[0] * (1.0f / 2048.0f) - mu * mu;
    const float rs  = rsqrtf(var + 1e-5f);

    unsigned hp[4], lp[4];
#pragma unroll
    for (int p = 0; p < 4; ++p) {
        float y0 = (v[2 * p]     - mu) * rs * gamma[base + 2 * p]     + beta[base + 2 * p];
        float y1 = (v[2 * p + 1] - mu) * rs * gamma[base + 2 * p + 1] + beta[base + 2 * p + 1];
        unsigned short h0 = bf16_rne(y0), h1 = bf16_rne(y1);
        unsigned short l0 = bf16_rne(y0 - bf16_to_f(h0));
        unsigned short l1 = bf16_rne(y1 - bf16_to_f(h1));
        hp[p] = (unsigned)h0 | ((unsigned)h1 << 16);
        lp[p] = (unsigned)l0 | ((unsigned)l1 << 16);
    }
    *reinterpret_cast<uint4v*>(hn_hi + (size_t)row * 2048 + base) = uint4v{hp[0], hp[1], hp[2], hp[3]};
    *reinterpret_cast<uint4v*>(hn_lo + (size_t)row * 2048 + base) = uint4v{lp[0], lp[1], lp[2], lp[3]};
}

// ===========================================================================
// FALLBACK PATH (small workspace): on-the-fly split, register-prefetch
// double buffering.
// ===========================================================================
__global__ __launch_bounds__(256) void gemm_nt_bf16x3(
    const float* __restrict__ A, const float* __restrict__ B,
    const float* __restrict__ bias, float* __restrict__ C,
    int M, int Nout, int K)
{
    __shared__ __align__(16) __bf16 As_hi[2][128][32];
    __shared__ __align__(16) __bf16 As_lo[2][128][32];
    __shared__ __align__(16) __bf16 Bs_hi[2][128][32];
    __shared__ __align__(16) __bf16 Bs_lo[2][128][32];

    const int t     = threadIdx.x;
    const int lane  = t & 31;
    const int w     = t >> 5;
    const int m0    = blockIdx.x * 128;
    const int n0    = blockIdx.y * 128;
    const int waveM = (w >> 2) * 64;
    const int waveN = (w & 3) * 32;
    const int rowL  = lane & 15;
    const int srow  = t >> 1;
    const int scol  = (t & 1) << 4;

    v8f acc[4][2];
#pragma unroll
    for (int j = 0; j < 2; ++j) {
        const int col = n0 + waveN + j * 16 + rowL;
        const float bz = (col < Nout) ? bias[col] : 0.0f;
        v8f init = {bz, bz, bz, bz, bz, bz, bz, bz};
#pragma unroll
        for (int i = 0; i < 4; ++i) acc[i][j] = init;
    }

    const float* aRow = A + (size_t)(m0 + srow) * K + scol;
    const bool   bOK  = (n0 + srow) < Nout;
    const float* bRow = B + (size_t)(bOK ? (n0 + srow) : 0) * K + scol;

    float avals[16], bvals[16];
#pragma unroll
    for (int i = 0; i < 16; ++i) bvals[i] = 0.0f;

    fetch16(aRow, avals);
    if (bOK) fetch16(bRow, bvals);
    split_store16(avals, &As_hi[0][srow][scol], &As_lo[0][srow][scol]);
    split_store16(bvals, &Bs_hi[0][srow][scol], &Bs_lo[0][srow][scol]);
    __syncthreads();

    const int nk = K >> 5;
    for (int kt = 0; kt < nk; ++kt) {
        const int  cur  = kt & 1;
        const bool more = (kt + 1) < nk;

        if (more) {
            const int k0 = (kt + 1) << 5;
            fetch16(aRow + k0, avals);
            if (bOK) fetch16(bRow + k0, bvals);
        }

        FragBF ah[4], al[4], bh[2], bl[2];
#pragma unroll
        for (int i = 0; i < 4; ++i) {
            const int r = waveM + i * 16 + rowL;
            ah[i] = load_fragA(As_hi[cur], r, lane);
            al[i] = load_fragA(As_lo[cur], r, lane);
        }
#pragma unroll
        for (int j = 0; j < 2; ++j) {
            const int r = waveN + j * 16 + rowL;
            bh[j] = load_fragB(Bs_hi[cur], r, lane);
            bl[j] = load_fragB(Bs_lo[cur], r, lane);
        }
#pragma unroll
        for (int i = 0; i < 4; ++i)
#pragma unroll
            for (int j = 0; j < 2; ++j) {
                acc[i][j] = __builtin_amdgcn_wmma_f32_16x16x32_bf16(
                    false, ah[i].v, false, bh[j].v, (short)0, acc[i][j], false, false);
                acc[i][j] = __builtin_amdgcn_wmma_f32_16x16x32_bf16(
                    false, al[i].v, false, bh[j].v, (short)0, acc[i][j], false, false);
                acc[i][j] = __builtin_amdgcn_wmma_f32_16x16x32_bf16(
                    false, ah[i].v, false, bl[j].v, (short)0, acc[i][j], false, false);
            }

        if (more) {
            const int nxt = cur ^ 1;
            split_store16(avals, &As_hi[nxt][srow][scol], &As_lo[nxt][srow][scol]);
            split_store16(bvals, &Bs_hi[nxt][srow][scol], &Bs_lo[nxt][srow][scol]);
        }
        __syncthreads();
    }

    const int rsel = (lane >> 4) << 3;
#pragma unroll
    for (int i = 0; i < 4; ++i)
#pragma unroll
        for (int j = 0; j < 2; ++j) {
            const int col = n0 + waveN + j * 16 + rowL;
            if (col < Nout) {
#pragma unroll
                for (int v = 0; v < 8; ++v) {
                    const int row = m0 + waveM + i * 16 + rsel + v;
                    C[(size_t)row * Nout + col] = acc[i][j][v];
                }
            }
        }
}

__global__ __launch_bounds__(256) void layernorm_2048(
    const float* __restrict__ h, const float* __restrict__ gamma,
    const float* __restrict__ beta, float* __restrict__ hn)
{
    const int row = blockIdx.x;
    const int t   = threadIdx.x;
    const float* hr = h + (size_t)row * 2048;

    float v[8];
    float s = 0.0f, ss = 0.0f;
#pragma unroll
    for (int i = 0; i < 8; ++i) {
        float x = hr[t + i * 256];
        v[i] = x;
        s  += x;
        ss += x * x;
    }

    __shared__ float r1[256], r2[256];
    r1[t] = s; r2[t] = ss;
    __syncthreads();
    for (int off = 128; off > 0; off >>= 1) {
        if (t < off) { r1[t] += r1[t + off]; r2[t] += r2[t + off]; }
        __syncthreads();
    }
    const float mu  = r1[0] * (1.0f / 2048.0f);
    const float var = r2[0] * (1.0f / 2048.0f) - mu * mu;
    const float rs  = rsqrtf(var + 1e-5f);

    float* o = hn + (size_t)row * 2048;
#pragma unroll
    for (int i = 0; i < 8; ++i) {
        const int idx = t + i * 256;
        o[idx] = (v[i] - mu) * rs * gamma[idx] + beta[idx];
    }
}

// ---------------------------------------------------------------------------
// Launch: GEMM1 (x @ w1^T + b1) -> LN -> GEMM2 (hn @ w3^T + b3)
// Fast path needs ws = h + hi/lo splits of {x, w1, hn, w3} = ~347 MB.
// Fallback needs 64 MB. Branch on ws_size is constant -> deterministic.
// ---------------------------------------------------------------------------
extern "C" void kernel_launch(void* const* d_in, const int* in_sizes, int n_in,
                              void* d_out, int out_size, void* d_ws, size_t ws_size,
                              hipStream_t stream) {
    (void)in_sizes; (void)n_in; (void)out_size;
    const float* x     = (const float*)d_in[0];  // [4096, 2048]
    const float* w1    = (const float*)d_in[1];  // [2048, 2048]
    const float* b1    = (const float*)d_in[2];  // [2048]
    const float* gamma = (const float*)d_in[3];  // [2048]
    const float* beta  = (const float*)d_in[4];  // [2048]
    const float* w3    = (const float*)d_in[5];  // [30000, 2048]
    const float* b3    = (const float*)d_in[6];  // [30000]
    float* logits = (float*)d_out;               // [4096, 30000]

    const int N = 4096, D = 2048, V = 30000;
    const size_t NB = (size_t)N * D;   // 8,388,608
    const size_t DB = (size_t)D * D;   // 4,194,304
    const size_t VB = (size_t)V * D;   // 61,440,000

    // fast-path workspace: h(f32) + bf16 hi/lo pairs for x, w1, hn, w3
    const size_t need = 4 * NB + 4 * NB + 4 * DB + 4 * NB + 4 * VB;

    if (ws_size >= need) {
        float*  h     = (float*)d_ws;                 // NB f32
        __bf16* x_hi  = (__bf16*)(h + NB);
        __bf16* x_lo  = x_hi + NB;
        __bf16* w1_hi = x_lo + NB;
        __bf16* w1_lo = w1_hi + DB;
        __bf16* hn_hi = w1_lo + DB;
        __bf16* hn_lo = hn_hi + NB;
        __bf16* w3_hi = hn_lo + NB;
        __bf16* w3_lo = w3_hi + VB;

        split_pair_kernel<<<(unsigned)((NB / 4 + 255) / 256), 256, 0, stream>>>(x,  x_hi,  x_lo,  (long)NB);
        split_pair_kernel<<<(unsigned)((DB / 4 + 255) / 256), 256, 0, stream>>>(w1, w1_hi, w1_lo, (long)DB);
        split_pair_kernel<<<(unsigned)((VB / 4 + 255) / 256), 256, 0, stream>>>(w3, w3_hi, w3_lo, (long)VB);

        gemm_nt_presplit<<<dim3(N / 128, D / 128), 256, 0, stream>>>(
            x_hi, x_lo, w1_hi, w1_lo, b1, h, N, D, D);
        layernorm_2048_split<<<dim3(N), 256, 0, stream>>>(h, gamma, beta, hn_hi, hn_lo);
        gemm_nt_presplit<<<dim3(N / 128, (V + 127) / 128), 256, 0, stream>>>(
            hn_hi, hn_lo, w3_hi, w3_lo, b3, logits, N, V, D);
    } else {
        float* h  = (float*)d_ws;
        float* hn = h + NB;
        gemm_nt_bf16x3<<<dim3(N / 128, D / 128), 256, 0, stream>>>(x, w1, b1, h, N, D, D);
        layernorm_2048<<<dim3(N), 256, 0, stream>>>(h, gamma, beta, hn);
        gemm_nt_bf16x3<<<dim3(N / 128, (V + 127) / 128), 256, 0, stream>>>(hn, w3, b3, logits, N, V, D);
    }
}
